// ConvolutionFeatureModel_40261023433362
// MI455X (gfx1250) — matrix operand order repaired and checked by reading, not compiled, not run on hardware
//
#include <hip/hip_runtime.h>
#include <math.h>

// ---------------------------------------------------------------------------
// relu(sqrt(max(||x||^2 + ||w||^2 - 2 x.w, 0)) / sqrt(512))
//   x: (8192, 512) f32   weight: (4096, 512) f32   out: (8192, 4096) f32
// Split-precision 3xBF16 GEMM on v_wmma_f32_16x16x32_bf16, double-buffered LDS.
// ---------------------------------------------------------------------------

typedef __attribute__((ext_vector_type(16))) __bf16 v16bf;
typedef __attribute__((ext_vector_type(8)))  __bf16 v8bf;
typedef __attribute__((ext_vector_type(4)))  __bf16 v4bf;
typedef __attribute__((ext_vector_type(8)))  float  v8f;
typedef __attribute__((ext_vector_type(4)))  float  v4f;

#define M_DIM 8192
#define N_DIM 4096
#define K_DIM 512
#define BM 128
#define BN 128
#define BK 32
#define BKP 40   // padded LDS row stride (elements) to stagger banks

// ---------------------------------------------------------------------------
// Row-norm precompute: one wave32 per row. Rows [0,8192) -> x, [8192,12288) -> w.
// ---------------------------------------------------------------------------
__global__ __launch_bounds__(256) void row_norms_kernel(
    const float* __restrict__ x, const float* __restrict__ w,
    float* __restrict__ norms) {
  const int gwave = (blockIdx.x * blockDim.x + threadIdx.x) >> 5;
  const int lane  = threadIdx.x & 31;
  const float* src;
  int row;
  if (gwave < M_DIM) { src = x; row = gwave; }
  else               { src = w; row = gwave - M_DIM; }
  const float* p = src + (size_t)row * K_DIM;
  float s = 0.0f;
#pragma unroll
  for (int k = 0; k < K_DIM / 32; ++k) {
    float v = p[lane + 32 * k];
    s = fmaf(v, v, s);
  }
#pragma unroll
  for (int off = 16; off > 0; off >>= 1) s += __shfl_xor(s, off, 32);
  if (lane == 0) norms[gwave] = s;
}

// ---------------------------------------------------------------------------
// fp32 -> (bf16 hi, bf16 lo) split
// ---------------------------------------------------------------------------
__device__ __forceinline__ void cvt_split4(const v4f f, v4bf& hi, v4bf& lo) {
#pragma unroll
  for (int e = 0; e < 4; ++e) {
    float v   = f[e];
    __bf16 h  = (__bf16)v;
    hi[e]     = h;
    lo[e]     = (__bf16)(v - (float)h);
  }
}

__device__ __forceinline__ v16bf cat8(v8bf a, v8bf b) {
  return __builtin_shufflevector(a, b, 0, 1, 2, 3, 4, 5, 6, 7,
                                       8, 9, 10, 11, 12, 13, 14, 15);
}

// ---------------------------------------------------------------------------
// Main GEMM + distance epilogue.
// Block: 256 threads = 8 waves (2 x 4 wave grid), 128x128 output tile.
// Wave: 64x32 patch = 4x2 tiles of 16x16; 3 WMMAs (hi*hi, hi*lo, lo*hi)
// per tile per BK=32 step. LDS double-buffered: one barrier per K-step.
// ---------------------------------------------------------------------------
__global__ __launch_bounds__(256) void dist_gemm_kernel(
    const float* __restrict__ x, const float* __restrict__ w,
    const float* __restrict__ norms, float* __restrict__ out) {
  __shared__ __align__(16) __bf16 As_hi[2][BM * BKP];
  __shared__ __align__(16) __bf16 As_lo[2][BM * BKP];
  __shared__ __align__(16) __bf16 Bs_hi[2][BN * BKP];
  __shared__ __align__(16) __bf16 Bs_lo[2][BN * BKP];

  const int tid  = threadIdx.x;
  const int lane = tid & 31;
  const int wave = tid >> 5;
  const int wm   = wave >> 2;   // 0..1  (M direction, 64 rows each)
  const int wn   = wave & 3;    // 0..3  (N direction, 32 cols each)

  const int blockM = blockIdx.y * BM;
  const int blockN = blockIdx.x * BN;

  // Global staging: thread loads one float4 per 32-row slab, 4 slabs/tile.
  const int lrow = tid >> 3;        // 0..31
  const int lk4  = (tid & 7) * 4;   // 0,4,...,28

  const float* aPtr = x + (size_t)(blockM + lrow) * K_DIM + lk4;
  const float* bPtr = w + (size_t)(blockN + lrow) * K_DIM + lk4;

  v8f acc[4][2] = {};

  v4f aReg[4], bReg[4];
  // Load + convert slab 0 into LDS buffer 0.
#pragma unroll
  for (int p = 0; p < 4; ++p) {
    aReg[p] = *(const v4f*)(aPtr + (size_t)(32 * p) * K_DIM);
    bReg[p] = *(const v4f*)(bPtr + (size_t)(32 * p) * K_DIM);
  }
#pragma unroll
  for (int p = 0; p < 4; ++p) {
    v4bf hi, lo;
    cvt_split4(aReg[p], hi, lo);
    *(v4bf*)&As_hi[0][(lrow + 32 * p) * BKP + lk4] = hi;
    *(v4bf*)&As_lo[0][(lrow + 32 * p) * BKP + lk4] = lo;
    cvt_split4(bReg[p], hi, lo);
    *(v4bf*)&Bs_hi[0][(lrow + 32 * p) * BKP + lk4] = hi;
    *(v4bf*)&Bs_lo[0][(lrow + 32 * p) * BKP + lk4] = lo;
  }
  // Prefetch slab 1 into registers.
#pragma unroll
  for (int p = 0; p < 4; ++p) {
    aReg[p] = *(const v4f*)(aPtr + (size_t)(32 * p) * K_DIM + BK);
    bReg[p] = *(const v4f*)(bPtr + (size_t)(32 * p) * K_DIM + BK);
  }
  __syncthreads();

  // Per-lane fragment geometry (matches CDNA5 bf16 WMMA VGPR layouts).
  const int lhalf = lane & 15;
  const int aK0   = (lane >> 4) * 8;    // A: segs {k0, k0+16}
  const int bK0   = (lane >> 4) * 8;    // B: segs {k0, k0+16}

  int buf = 0;
  for (int kk = 0; kk < K_DIM; kk += BK) {
    // B fragments for this wave's two 16-col tiles (current buffer).
    v16bf Bhi[2], Blo[2];
#pragma unroll
    for (int j = 0; j < 2; ++j) {
      const int nloc = wn * 32 + j * 16 + lhalf;
      const __bf16* pbh = &Bs_hi[buf][nloc * BKP + bK0];
      const __bf16* pbl = &Bs_lo[buf][nloc * BKP + bK0];
      Bhi[j] = cat8(*(const v8bf*)pbh, *(const v8bf*)(pbh + 16));
      Blo[j] = cat8(*(const v8bf*)pbl, *(const v8bf*)(pbl + 16));
    }
    v16bf Ahi[4], Alo[4];
#pragma unroll
    for (int t = 0; t < 4; ++t) {
      const int mloc = wm * 64 + t * 16 + lhalf;
      const __bf16* pah = &As_hi[buf][mloc * BKP + aK0];
      const __bf16* pal = &As_lo[buf][mloc * BKP + aK0];
      Ahi[t] = cat8(*(const v8bf*)pah, *(const v8bf*)(pah + 16));
      Alo[t] = cat8(*(const v8bf*)pal, *(const v8bf*)(pal + 16));
    }

    // Fill the other buffer for the next step (overlaps with WMMA burst),
    // then deepen the global prefetch by one more slab.
    if (kk + BK < K_DIM) {
      const int nb = buf ^ 1;
#pragma unroll
      for (int p = 0; p < 4; ++p) {
        v4bf hi, lo;
        cvt_split4(aReg[p], hi, lo);
        *(v4bf*)&As_hi[nb][(lrow + 32 * p) * BKP + lk4] = hi;
        *(v4bf*)&As_lo[nb][(lrow + 32 * p) * BKP + lk4] = lo;
        cvt_split4(bReg[p], hi, lo);
        *(v4bf*)&Bs_hi[nb][(lrow + 32 * p) * BKP + lk4] = hi;
        *(v4bf*)&Bs_lo[nb][(lrow + 32 * p) * BKP + lk4] = lo;
      }
      if (kk + 2 * BK < K_DIM) {
        const int koff = kk + 2 * BK;
#pragma unroll
        for (int p = 0; p < 4; ++p) {
          aReg[p] = *(const v4f*)(aPtr + (size_t)(32 * p) * K_DIM + koff);
          bReg[p] = *(const v4f*)(bPtr + (size_t)(32 * p) * K_DIM + koff);
        }
      }
    }

#pragma unroll
    for (int t = 0; t < 4; ++t) {
#pragma unroll
      for (int j = 0; j < 2; ++j) {
        acc[t][j] = __builtin_amdgcn_wmma_f32_16x16x32_bf16(
            false, Ahi[t], false, Bhi[j], (short)0, acc[t][j], false, false);
        acc[t][j] = __builtin_amdgcn_wmma_f32_16x16x32_bf16(
            false, Ahi[t], false, Blo[j], (short)0, acc[t][j], false, false);
        acc[t][j] = __builtin_amdgcn_wmma_f32_16x16x32_bf16(
            false, Alo[t], false, Bhi[j], (short)0, acc[t][j], false, false);
      }
    }

    __syncthreads();   // next-buffer stores visible; this-buffer reads done
    buf ^= 1;
  }

  // Epilogue: dist = sqrt(max(|x|^2 + |w|^2 - 2 dot, 0)) / 22.627417
  const float inv_scale = 1.0f / 22.627417f;
  const float* xsq = norms;
  const float* wsq = norms + M_DIM;
  const int mAdd = (lane >> 4) << 3;   // +8 for upper half-wave

#pragma unroll
  for (int t = 0; t < 4; ++t) {
#pragma unroll
    for (int j = 0; j < 2; ++j) {
      const int gmBase = blockM + wm * 64 + t * 16 + mAdd;
      const int gn     = blockN + wn * 32 + j * 16 + lhalf;
      const float wn2  = wsq[gn];
#pragma unroll
      for (int r = 0; r < 8; ++r) {
        const int gm = gmBase + r;
        float d2 = xsq[gm] + wn2 - 2.0f * acc[t][j][r];
        d2 = fmaxf(d2, 0.0f);
        out[(size_t)gm * N_DIM + gn] = sqrtf(d2) * inv_scale;
      }
    }
  }
}

// ---------------------------------------------------------------------------
extern "C" void kernel_launch(void* const* d_in, const int* in_sizes, int n_in,
                              void* d_out, int out_size, void* d_ws, size_t ws_size,
                              hipStream_t stream) {
  const float* x = (const float*)d_in[0];
  const float* w = (const float*)d_in[1];
  float* out   = (float*)d_out;
  float* norms = (float*)d_ws;   // 12288 floats = 48 KB

  // 12288 rows, one wave each, 8 waves per block.
  row_norms_kernel<<<(M_DIM + N_DIM) / 8, 256, 0, stream>>>(x, w, norms);

  dim3 grid(N_DIM / BN, M_DIM / BM);   // (32, 64)
  dist_gemm_kernel<<<grid, 256, 0, stream>>>(x, w, norms, out);
}
